// MultiHeadSelfAttention_26688926777453
// MI455X (gfx1250) — matrix operand order repaired and checked by reading, not compiled, not run on hardware
//
#include <hip/hip_runtime.h>

typedef __attribute__((ext_vector_type(16))) __bf16 v16bf;
typedef __attribute__((ext_vector_type(8)))  float  v8f;

union Frag {
    uint4 q[2];
    v16bf v;
};

__device__ __forceinline__ unsigned short bfbits(float f) {
    union { __bf16 h; unsigned short u; } cv;
    cv.h = (__bf16)f;                 // RNE f32->bf16 (hw cvt)
    return cv.u;
}

#define N_TOK 4096
#define C_DIM 256
#define HEADS 8
#define DHEAD 32
#define BATCH 2

// async global->LDS copy of 16B per lane (ASYNCcnt-tracked)
#define ASYNC_B128(ldsoff, gaddr) \
    asm volatile("global_load_async_to_lds_b128 %0, %1, off" :: "v"(ldsoff), "v"(gaddr) : "memory")
#define WAIT_ASYNC0() asm volatile("s_wait_asynccnt 0x0" ::: "memory")

// ---------------------------------------------------------------------------
// Kernel 1a: tokens[b][i][ch] = bf16( x[b][ch][i] + PE(i, ch) )
// ---------------------------------------------------------------------------
__global__ __launch_bounds__(256) void prep_tokens(const float* __restrict__ x,
                                                   unsigned short* __restrict__ tok) {
    int t  = blockIdx.x * 256 + threadIdx.x;  // t = bi*2^20 + i*256 + ch
    int ch = t & 255;
    int i  = (t >> 8) & 4095;
    int bi = t >> 20;
    int chh = ch & 127;
    float rate = __expf((float)chh * -0.07195578415606394f); // -ln(10000)/128
    float ang  = (float)i * rate;
    float pe   = (ch < 128) ? __sinf(ang) : __cosf(ang);
    float val  = x[((size_t)bi * C_DIM + ch) * N_TOK + i] + pe;
    tok[t] = bfbits(val);
}

// ---------------------------------------------------------------------------
// Kernel 1b: bf16 copy of W_qkv [768, 256]
// ---------------------------------------------------------------------------
__global__ __launch_bounds__(256) void prep_w(const float* __restrict__ w,
                                              unsigned short* __restrict__ wb) {
    int t = blockIdx.x * 256 + threadIdx.x;
    if (t < 768 * 256) wb[t] = bfbits(w[t]);
}

// ---------------------------------------------------------------------------
// Kernel 2: QKV GEMM.  C[i, j] = sum_k tokens[i,k] * W[j,k]  (per batch)
// Q,K scattered to [bh][n][32] bf16 (Q pre-scaled by d^-0.5).
// V chunk-blocked: [bh][blk=n/32][dd=32][key=32] bf16 so each 32-key chunk is
// a contiguous 2KB region (linear async-copy source for the attention kernel).
// ---------------------------------------------------------------------------
__global__ __launch_bounds__(128) void qkv_gemm(const unsigned short* __restrict__ tok,
                                                const unsigned short* __restrict__ wb,
                                                unsigned short* __restrict__ qws,
                                                unsigned short* __restrict__ kws,
                                                unsigned short* __restrict__ vws) {
    int lane = threadIdx.x & 31;
    int hl = lane & 15, half = lane >> 4;
    int gw  = blockIdx.x * 4 + (threadIdx.x >> 5);   // 0..24575
    int bi  = gw / 12288;
    int rem = gw - bi * 12288;
    int tm  = rem / 48;          // 0..255  (row tile over n)
    int tn  = rem - tm * 48;     // 0..47   (col tile over 3c)

    const unsigned short* arow = tok + (size_t)bi * N_TOK * C_DIM + (size_t)(tm * 16 + hl) * C_DIM;
    const unsigned short* brow = wb + (size_t)(tn * 16 + hl) * C_DIM;

    v8f acc = {};
    for (int kb = 0; kb < 8; ++kb) {
        Frag a, b;
        // A frag: row = tm*16 + lane%16; K = 16*half..+15 contiguous
        const uint4* ap = (const uint4*)(arow + kb * 32);
        a.q[0] = ap[2 * half];
        a.q[1] = ap[2 * half + 1];
        // B frag: column j = tn*16 + lane%16 (= W row); K = 16*half..+15 contiguous
        const uint4* bp = (const uint4*)(brow + kb * 32 + 16 * half);
        b.q[0] = bp[0];
        b.q[1] = bp[1];
        acc = __builtin_amdgcn_wmma_f32_16x16x32_bf16(false, a.v, false, b.v,
                                                      (short)0, acc, false, false);
    }

    // C layout: VGPR r, lanes 0-15 -> row r, lanes 16-31 -> row r+8; col = lane%16
    int jcol = tn * 16 + hl;     // 0..767
    int seg  = jcol >> 8;        // 0=q 1=k 2=v (uniform per wave)
    int jc   = jcol & 255;
    int head = jc >> 5;
    int dd   = jc & 31;
    int rowbase = tm * 16 + 8 * half;
    size_t hb = (size_t)bi * HEADS + head;
    if (seg == 0) {
        unsigned short* qp = qws + hb * N_TOK * DHEAD + dd;
        #pragma unroll
        for (int r = 0; r < 8; ++r)
            qp[(size_t)(rowbase + r) * DHEAD] = bfbits(acc[r] * 0.17677669529663687f);
    } else if (seg == 1) {
        unsigned short* kp = kws + hb * N_TOK * DHEAD + dd;
        #pragma unroll
        for (int r = 0; r < 8; ++r)
            kp[(size_t)(rowbase + r) * DHEAD] = bfbits(acc[r]);
    } else {
        unsigned short tmp[8];
        #pragma unroll
        for (int r = 0; r < 8; ++r) tmp[r] = bfbits(acc[r]);
        // chunk-blocked V^T: element (token i, dd) -> [(hb*128 + i/32)*32 + dd]*32 + (i%32)
        unsigned short* vp = vws + ((hb * 128 + (rowbase >> 5)) * 32 + dd) * 32 + (rowbase & 31);
        *(uint4*)vp = *(const uint4*)tmp;   // 8 consecutive token positions
    }
}

// ---------------------------------------------------------------------------
// Kernel 3: flash attention. 512 threads = 16 waves, all on one (batch, head);
// wave w owns query block qg*16+w. 32-key K/V chunks are double-buffered in
// LDS via global_load_async_to_lds_b128 (wave 0 issues), padded to 80B rows
// for conflict-free half-wave reads. 2 QK + 2 PV wmmas per iteration.
// ---------------------------------------------------------------------------
__global__ __launch_bounds__(512) void attn(const unsigned short* __restrict__ qws,
                                            const unsigned short* __restrict__ kws,
                                            const unsigned short* __restrict__ vws,
                                            float* __restrict__ out) {
    // per buffer: K = 32 rows x 80B (2560B), then V = 32 rows x 80B (2560B)
    __shared__ __align__(16) unsigned short kvbuf[2][2 * 32 * 40];
    __shared__ __align__(16) unsigned short pst[16][16 * 40];   // padded P staging

    int wave = threadIdx.x >> 5;
    int lane = threadIdx.x & 31;
    int hl = lane & 15, half = lane >> 4;

    int bh = blockIdx.x >> 4;            // 0..15 : bi*8 + head
    int qg = blockIdx.x & 15;
    int qb = qg * 16 + wave;             // 0..255
    int head = bh & 7;
    int bi   = bh >> 3;

    const unsigned short* qbase = qws + (size_t)bh * N_TOK * DHEAD;
    const unsigned short* kbase = kws + (size_t)bh * N_TOK * DHEAD;
    const unsigned short* vbase = vws + (size_t)bh * 128 * 1024;

    Frag qa;  // Q tile 16x32 (already scaled by d^-0.5)
    {
        const uint4* qp = (const uint4*)(qbase + (size_t)(qb * 16 + hl) * DHEAD);
        qa.q[0] = qp[2 * half];
        qa.q[1] = qp[2 * half + 1];
    }

    // per-lane padded LDS dest offsets for the 4 b128 copies of a 2KB chunk:
    // chunk = t*32+lane; row = chunk>>2 (80B stride), part = chunk&3 (16B)
    unsigned ldsoff[4];
    #pragma unroll
    for (int t = 0; t < 4; ++t) {
        int chunk = t * 32 + lane;
        ldsoff[t] = (unsigned)((chunk >> 2) * 80 + (chunk & 3) * 16);
    }
    unsigned kvbase0 = (unsigned)(size_t)&kvbuf[0][0];   // low 32 bits = LDS offset
    unsigned kvbase1 = (unsigned)(size_t)&kvbuf[1][0];
    const unsigned VOFF = 32 * 80;                        // V region inside a buffer

    // per-lane global source addresses (wave 0 only issues)
    unsigned long long kga = (unsigned long long)(size_t)kbase + (unsigned long long)(lane * 16);
    unsigned long long vga = (unsigned long long)(size_t)vbase + (unsigned long long)(lane * 16);

    if (wave == 0) {  // prologue: fill buffer 0 with chunk 0
        #pragma unroll
        for (int t = 0; t < 4; ++t) {
            ASYNC_B128(kvbase0 + ldsoff[t],        kga + (unsigned long long)(t * 512));
            ASYNC_B128(kvbase0 + VOFF + ldsoff[t], vga + (unsigned long long)(t * 512));
        }
        WAIT_ASYNC0();
    }
    __syncthreads();

    v8f O0 = {}, O1 = {};
    float rm[8], rl[8];
    #pragma unroll
    for (int r = 0; r < 8; ++r) { rm[r] = -3.0e38f; rl[r] = 0.0f; }

    unsigned short* myp = &pst[wave][0];

    for (int it = 0; it < 128; ++it) {
        const unsigned short* cb = &kvbuf[it & 1][0];
        if (wave == 0 && it + 1 < 128) {   // prefetch next chunk into other buffer
            unsigned nb = (it & 1) ? kvbase0 : kvbase1;
            unsigned long long ko = kga + (unsigned long long)(it + 1) * 2048ull;
            unsigned long long vo = vga + (unsigned long long)(it + 1) * 2048ull;
            #pragma unroll
            for (int t = 0; t < 4; ++t) {
                ASYNC_B128(nb + ldsoff[t],        ko + (unsigned long long)(t * 512));
                ASYNC_B128(nb + VOFF + ldsoff[t], vo + (unsigned long long)(t * 512));
            }
        }

        // B frags from LDS (80B row stride -> conflict-free across a half-wave)
        Frag k0, k1, v0, v1;
        {
            const uint4* kp0 = (const uint4*)(cb + hl * 40 + 16 * half);
            k0.q[0] = kp0[0]; k0.q[1] = kp0[1];
            const uint4* kp1 = (const uint4*)(cb + (16 + hl) * 40 + 16 * half);
            k1.q[0] = kp1[0]; k1.q[1] = kp1[1];
            const uint4* vp0 = (const uint4*)(cb + 32 * 40 + hl * 40 + 16 * half);
            v0.q[0] = vp0[0]; v0.q[1] = vp0[1];
            const uint4* vp1 = (const uint4*)(cb + 32 * 40 + (16 + hl) * 40 + 16 * half);
            v1.q[0] = vp1[0]; v1.q[1] = vp1[1];
        }

        v8f S0 = {}, S1 = {};
        S0 = __builtin_amdgcn_wmma_f32_16x16x32_bf16(false, qa.v, false, k0.v, (short)0, S0, false, false);
        S1 = __builtin_amdgcn_wmma_f32_16x16x32_bf16(false, qa.v, false, k1.v, (short)0, S1, false, false);

        // online softmax: row r+8*half lives across the 16 lanes of a half-wave
        #pragma unroll
        for (int r = 0; r < 8; ++r) {
            float mx = fmaxf(S0[r], S1[r]);
            mx = fmaxf(mx, __shfl_xor(mx, 1));
            mx = fmaxf(mx, __shfl_xor(mx, 2));
            mx = fmaxf(mx, __shfl_xor(mx, 4));
            mx = fmaxf(mx, __shfl_xor(mx, 8));
            float mnew  = fmaxf(rm[r], mx);
            float alpha = __expf(rm[r] - mnew);
            rm[r] = mnew;
            float e0 = __expf(S0[r] - mnew);
            float e1 = __expf(S1[r] - mnew);
            rl[r] = rl[r] * alpha + e0 + e1;   // per-lane partial; reduced at end
            O0[r] *= alpha;
            O1[r] *= alpha;
            myp[(r + 8 * half) * 40 + hl]      = bfbits(e0);
            myp[(r + 8 * half) * 40 + 16 + hl] = bfbits(e1);
        }

        // read P back in A-matrix layout (same-wave LDS ops are in-order)
        Frag pa;
        const uint4* pp = (const uint4*)(myp + hl * 40);
        pa.q[0] = pp[2 * half];
        pa.q[1] = pp[2 * half + 1];

        O0 = __builtin_amdgcn_wmma_f32_16x16x32_bf16(false, pa.v, false, v0.v, (short)0, O0, false, false);
        O1 = __builtin_amdgcn_wmma_f32_16x16x32_bf16(false, pa.v, false, v1.v, (short)0, O1, false, false);

        if (wave == 0 && it + 1 < 128) WAIT_ASYNC0();
        __syncthreads();   // next buffer ready / this buffer reusable
    }

    #pragma unroll
    for (int r = 0; r < 8; ++r) {
        float l = rl[r];
        l += __shfl_xor(l, 1);
        l += __shfl_xor(l, 2);
        l += __shfl_xor(l, 4);
        l += __shfl_xor(l, 8);
        float inv = 1.0f / l;
        O0[r] *= inv;
        O1[r] *= inv;
    }

    // out[b][head*32+dd][qrow] ; per lane 8 consecutive qrows -> 2x float4
    size_t ch0 = (size_t)bi * C_DIM + head * DHEAD + hl;
    size_t ch1 = ch0 + 16;
    int qrow = qb * 16 + 8 * half;
    float* o0p = out + ch0 * N_TOK + qrow;
    float* o1p = out + ch1 * N_TOK + qrow;
    *(float4*)(o0p)     = make_float4(O0[0], O0[1], O0[2], O0[3]);
    *(float4*)(o0p + 4) = make_float4(O0[4], O0[5], O0[6], O0[7]);
    *(float4*)(o1p)     = make_float4(O1[0], O1[1], O1[2], O1[3]);
    *(float4*)(o1p + 4) = make_float4(O1[4], O1[5], O1[6], O1[7]);
}

// ---------------------------------------------------------------------------
extern "C" void kernel_launch(void* const* d_in, const int* in_sizes, int n_in,
                              void* d_out, int out_size, void* d_ws, size_t ws_size,
                              hipStream_t stream) {
    const float* x = (const float*)d_in[0];   // [2,256,64,64]
    const float* w = (const float*)d_in[1];   // [768,256]
    float* out = (float*)d_out;               // [2,256,64,64]

    char* ws = (char*)d_ws;
    unsigned short* tok = (unsigned short*)(ws);             // 4 MiB  tokens bf16 [b][n][c]
    unsigned short* wb  = (unsigned short*)(ws + 4194304);   // 384 KiB W bf16
    unsigned short* qws = (unsigned short*)(ws + 4587520);   // 4 MiB  Q [bh][n][32]
    unsigned short* kws = (unsigned short*)(ws + 8781824);   // 4 MiB  K [bh][n][32]
    unsigned short* vws = (unsigned short*)(ws + 12976128);  // 4 MiB  V [bh][blk][32][32]

    prep_tokens<<<8192, 256, 0, stream>>>(x, tok);
    prep_w<<<768, 256, 0, stream>>>(w, wb);
    qkv_gemm<<<6144, 128, 0, stream>>>(tok, wb, qws, kws, vws);
    attn<<<256, 512, 0, stream>>>(qws, kws, vws, out);
}
